// Attention_Rel_Vec_5643587027258
// MI455X (gfx1250) — matrix operand order repaired and checked by reading, not compiled, not run on hardware
//
#include <hip/hip_runtime.h>

// ---------------------------------------------------------------------------
// Music-Transformer-style relative attention, MI455X (gfx1250), wave32 WMMA.
// B=2, S=2048, D=1024, H=16, hd=64.  All matmuls via v_wmma_f32_16x16x32_bf16.
// Flash-attention structure; the relative-position "skew" is realized by
// computing a 65x2048 bf16 QEr slab per 64-row block in LDS and gathering
// shifted entries per score element.  Q/K/V tiles are staged global->LDS with
// the CDNA5 Tensor Data Mover (tensor_load_to_lds, TENSORcnt), K/V double-
// buffered so DMA overlaps WMMA compute.
// ---------------------------------------------------------------------------

#define S_LEN  2048
#define DMODEL 1024
#define NH     16
#define HD     64
#define BQ     64      // query rows per workgroup (flash)
#define BKV    32      // key/value columns per inner iteration

typedef __bf16 bf16x16 __attribute__((ext_vector_type(16)));
typedef __bf16 bf16x8  __attribute__((ext_vector_type(8)));
typedef float  f32x8   __attribute__((ext_vector_type(8)));
typedef unsigned int u32x4 __attribute__((ext_vector_type(4)));
typedef int          i32x4 __attribute__((ext_vector_type(4)));
typedef int          i32x8 __attribute__((ext_vector_type(8)));

union U8  { bf16x8  v; unsigned short u[8];  };
union U16 { bf16x16 v; unsigned short u[16]; };

__device__ inline unsigned short f2bf(float f) {           // RNE float->bf16
  unsigned int u = __builtin_bit_cast(unsigned int, f);
  u += 0x7FFFu + ((u >> 16) & 1u);
  return (unsigned short)(u >> 16);
}
__device__ inline float bf2f(unsigned short h) {
  unsigned int u = ((unsigned int)h) << 16;
  return __builtin_bit_cast(float, u);
}

__device__ inline f32x8 wmma_bf16(bf16x16 a, bf16x16 b, f32x8 c) {
  // emits v_wmma_f32_16x16x32_bf16
  return __builtin_amdgcn_wmma_f32_16x16x32_bf16(false, a, false, b,
                                                 (short)0, c, false, false);
}

// A-fragment (16x32, bf16): lane M = lane%16; K = 16*hi+{0..7} and 16*hi+8+{0..7}
__device__ inline bf16x16 ldA(const unsigned short* p_lo, const unsigned short* p_hi) {
  bf16x8 lo = *(const bf16x8*)p_lo;
  bf16x8 hi = *(const bf16x8*)p_hi;
  bf16x16 r;
#pragma unroll
  for (int i = 0; i < 8; ++i) { r[i] = lo[i]; r[i + 8] = hi[i]; }
  return r;
}
// B-fragment (32x16, bf16): lane N = lane%16; K = 16*hi + {0..15} contiguous
__device__ inline bf16x16 ldB(const unsigned short* p) {
  return *(const bf16x16*)p;
}
// B-fragment with strided K (row-major [k][n] tile): element i at p[i*stride]
__device__ inline bf16x16 ldBstride(const unsigned short* p, int stride) {
  U16 r;
#pragma unroll
  for (int i = 0; i < 16; ++i) r.u[i] = p[i * stride];
  return r.v;
}

// ---------------------------------------------------------------------------
// Tensor Data Mover: 2-D tile, 2-byte elements, global -> LDS.
// D# packing per CDNA5 ISA §8.3/8.4 (group0: count/lds/global/type,
// group1: data_size, tensor_dim0/1, tile_dim0/1, dim0 stride).  Groups 2/3
// zero (<=2D tensor).  Rows past tensor_dim1 read as zero (OOB rule).
// This toolchain exposes the 6-argument builtin form.
// ---------------------------------------------------------------------------
__device__ inline void tdm_load_2d(unsigned lds_byte_off, const void* gaddr,
                                   unsigned tile_rows, unsigned row_halfs,
                                   unsigned tensor_rows, unsigned stride_halfs) {
  unsigned long long ga = (unsigned long long)gaddr;
  u32x4 g0;
  g0[0] = 1u;                                    // count=1, user descriptor
  g0[1] = lds_byte_off;                          // lds_addr (bytes)
  g0[2] = (unsigned)(ga & 0xFFFFFFFFu);          // global_addr[31:0]
  g0[3] = (unsigned)((ga >> 32) & 0x01FFFFFFu) | 0x80000000u;  // [56:32] | type=2
  i32x8 g1;
  g1[0] = 1 << 16;                               // data_size=1 (2 bytes)
  g1[1] = (int)(row_halfs << 16);                // tensor_dim0[15:0] @ bits 63:48
  g1[2] = (int)((row_halfs >> 16) | (tensor_rows << 16));  // dim0 hi | dim1 lo
  g1[3] = (int)((tensor_rows >> 16) | (row_halfs << 16));  // dim1 hi | tile_dim0
  g1[4] = (int)(tile_rows & 0xFFFFu);            // tile_dim1 (tile_dim2 = 0)
  g1[5] = (int)stride_halfs;                     // tensor_dim0_stride[31:0]
  g1[6] = 0;                                     // stride hi | dim1_stride lo
  g1[7] = 0;
  i32x4 gz4 = {0, 0, 0, 0};
  i32x8 gz8 = {0, 0, 0, 0, 0, 0, 0, 0};
  __builtin_amdgcn_tensor_load_to_lds(g0, g1, gz4, gz4, gz8, 0);
}

// ---------------------------------------------------------------------------
// cast fp32 -> bf16 (flat)
// ---------------------------------------------------------------------------
__global__ void k_cast(const float* __restrict__ in, unsigned short* __restrict__ out, int n) {
  int i = blockIdx.x * 256 + threadIdx.x;
  if (i < n) out[i] = f2bf(in[i]);
}

// ---------------------------------------------------------------------------
// QKV projection: y = x @ W^T, [4096,1024] x [1024,1024] bf16 WMMA GEMM,
// fp32 accumulate.  y[m,n] = sum_k x[m,k] * W[n,k]: W's native row-major
// [n][k] layout is already per-lane-contiguous-K for the B fragment, so both
// LDS tiles stage as straight b128 copies (no transpose anywhere).
// Output scattered to head-major [b,h,s,hd] bf16.
// grid (16, 64, 3), block 128 (4 waves); each wave does a 16x64 output strip.
// ---------------------------------------------------------------------------
__global__ __launch_bounds__(128) void k_qkv(
    const unsigned short* __restrict__ xb,
    const unsigned short* __restrict__ Wb3,
    unsigned short* __restrict__ Out3) {
  __shared__ __align__(32) unsigned short As[64 * 32];   // [m][k]
  __shared__ __align__(32) unsigned short Bs[64 * 32];   // [n][k]

  const int n0 = blockIdx.x * 64;
  const int m0 = blockIdx.y * 64;
  const unsigned short* W = Wb3 + (size_t)blockIdx.z * DMODEL * DMODEL;
  unsigned short* O = Out3 + (size_t)blockIdx.z * 2 * NH * S_LEN * HD;

  const int tid = threadIdx.x, wave = tid >> 5, lane = tid & 31;
  const int lr = lane & 15, hi = lane >> 4;

  const f32x8 zero = {0.f, 0.f, 0.f, 0.f, 0.f, 0.f, 0.f, 0.f};
  f32x8 acc[4];
#pragma unroll
  for (int n = 0; n < 4; ++n) acc[n] = zero;

  const int srow = tid >> 1, scc = (tid & 1) * 16;   // staging coords (64x32)
  for (int k0 = 0; k0 < DMODEL; k0 += 32) {
    __syncthreads();
    // stage A tile [m][k] and B tile [n][k]: both straight 16-half copies
    *(bf16x16*)(As + srow * 32 + scc) =
        *(const bf16x16*)(xb + (size_t)(m0 + srow) * DMODEL + k0 + scc);
    *(bf16x16*)(Bs + srow * 32 + scc) =
        *(const bf16x16*)(W + (size_t)(n0 + srow) * DMODEL + k0 + scc);
    __syncthreads();
    const int arow = wave * 16 + lr;
    bf16x16 a = ldA(As + arow * 32 + 16 * hi, As + arow * 32 + 8 + 16 * hi);
#pragma unroll
    for (int n = 0; n < 4; ++n) {
      bf16x16 b = ldB(Bs + (n * 16 + lr) * 32 + 16 * hi);
      acc[n] = wmma_bf16(a, b, acc[n]);
    }
  }
  // scatter to [b,h,s,hd]
#pragma unroll
  for (int n = 0; n < 4; ++n) {
    int nn = n0 + n * 16 + lr;
    int hh = nn >> 6, d = nn & 63;
#pragma unroll
    for (int i = 0; i < 8; ++i) {
      int m = m0 + wave * 16 + i + 8 * hi;
      int bb = m >> 11, s = m & 2047;
      O[(((size_t)bb * NH + hh) * S_LEN + s) * HD + d] = f2bf(acc[n][i]);
    }
  }
}

// ---------------------------------------------------------------------------
// Flash attention with relative-position skew.
// grid (S/64, H, B), block 128 (4 waves, wave w owns query rows w*16..w*16+15).
// Dynamic LDS 296,960 B < 320KB WGP LDS (offsets in halfs):
//   Gl [65][2048] @0       QEr slab rows s0..s0+64
//   Qs [80][64]   @133120  Q rows s0..s0+79 (TDM; OOB rows read as 0)
//   K0/K1 [32][64]@138240/140288  K tile [t][k], double buffered (TDM)
//   V0/V1 [32][64]@142336/144384  V tile [t][d], double buffered (TDM)
//   Ps [64][32]   @146432  probability tile
// ---------------------------------------------------------------------------
#define LDS_G  0u
#define LDS_Q  133120u
#define LDS_K0 138240u
#define LDS_K1 140288u
#define LDS_V0 142336u
#define LDS_V1 144384u
#define LDS_P  146432u
#define LDS_TOTAL_HALFS 148480u

__global__ __launch_bounds__(128) void k_flash(
    const unsigned short* __restrict__ Qb,
    const unsigned short* __restrict__ Kb,
    const unsigned short* __restrict__ Vb,
    const unsigned short* __restrict__ Erb,
    float* __restrict__ Obuf) {
  extern __shared__ unsigned short smem[];
  unsigned short* Gl = smem + LDS_G;
  unsigned short* Qs = smem + LDS_Q;
  unsigned short* Ps = smem + LDS_P;

  const int s0 = blockIdx.x * BQ;
  const int h  = blockIdx.y;
  const int b  = blockIdx.z;
  const int tid = threadIdx.x, wave = tid >> 5, lane = tid & 31;
  const int lr = lane & 15, hi = lane >> 4;

  const size_t head = ((size_t)b * NH + h) * S_LEN;
  const unsigned short* Qg = Qb + head * HD;
  const unsigned short* Kg = Kb + head * HD;
  const unsigned short* Vg = Vb + head * HD;

  // ---- TDM prologue: Q slab (80 rows, OOB rows -> 0) + first K/V tiles ----
  if (wave == 0) {
    tdm_load_2d(LDS_Q * 2,  Qg + (size_t)s0 * HD, 80, HD, (unsigned)(S_LEN - s0), HD);
    tdm_load_2d(LDS_K0 * 2, Kg,                   BKV, HD, BKV, HD);
    tdm_load_2d(LDS_V0 * 2, Vg,                   BKV, HD, BKV, HD);
    __builtin_amdgcn_s_wait_tensorcnt(0);
  }
  __syncthreads();

  // ---- phase 1: QEr slab  G[r][j] = q[s0+r] . Er[j], rows 0..64 kept ----
  for (int nt = wave; nt < S_LEN / 16; nt += 4) {
    int j0 = nt * 16;
    bf16x16 b0 = ldB(Erb + (size_t)(j0 + lr) * HD + 16 * hi);
    bf16x16 b1 = ldB(Erb + (size_t)(j0 + lr) * HD + 32 + 16 * hi);
#pragma unroll
    for (int mt = 0; mt < 5; ++mt) {
      int row = mt * 16 + lr;
      bf16x16 a0 = ldA(Qs + row * 64 + 16 * hi,      Qs + row * 64 + 8 + 16 * hi);
      bf16x16 a1 = ldA(Qs + row * 64 + 32 + 16 * hi, Qs + row * 64 + 40 + 16 * hi);
      f32x8 acc = {0.f, 0.f, 0.f, 0.f, 0.f, 0.f, 0.f, 0.f};
      acc = wmma_bf16(a0, b0, acc);
      acc = wmma_bf16(a1, b1, acc);
#pragma unroll
      for (int i = 0; i < 8; ++i) {
        int r = mt * 16 + i + 8 * hi;
        if (r < 65) Gl[r * 2048 + j0 + lr] = f2bf(acc[i]);
      }
    }
  }
  __syncthreads();

  // ---- phase 2: flash loop over key/value column tiles ----
  const float scale = 0.03125f;   // D^-0.5 = 1/32
  float mrow[8], lrow[8];
  f32x8 Oacc[4];
  const f32x8 zero = {0.f, 0.f, 0.f, 0.f, 0.f, 0.f, 0.f, 0.f};
#pragma unroll
  for (int i = 0; i < 8; ++i) { mrow[i] = -3.0e38f; lrow[i] = 0.f; }
#pragma unroll
  for (int n = 0; n < 4; ++n) Oacc[n] = zero;

  const int qrow = wave * 16 + lr;
  bf16x16 aq0 = ldA(Qs + qrow * 64 + 16 * hi,      Qs + qrow * 64 + 8 + 16 * hi);
  bf16x16 aq1 = ldA(Qs + qrow * 64 + 32 + 16 * hi, Qs + qrow * 64 + 40 + 16 * hi);

  const int NIT = S_LEN / BKV;
  for (int it = 0; it < NIT; ++it) {
    const int t0 = it * BKV;
    const int cur = it & 1;
    // issue DMA for the next tile pair into the other buffer (overlaps compute)
    if (wave == 0 && it + 1 < NIT) {
      int tn = t0 + BKV;
      tdm_load_2d((cur ? LDS_K0 : LDS_K1) * 2, Kg + (size_t)tn * HD, BKV, HD, BKV, HD);
      tdm_load_2d((cur ? LDS_V0 : LDS_V1) * 2, Vg + (size_t)tn * HD, BKV, HD, BKV, HD);
    }
    const unsigned short* Kr = smem + (cur ? LDS_K1 : LDS_K0);
    const unsigned short* Vr = smem + (cur ? LDS_V1 : LDS_V0);

    // S = Q K^T (two 16x16 n-tiles per wave)
    f32x8 st[2];
#pragma unroll
    for (int n = 0; n < 2; ++n) {
      int trow = n * 16 + lr;
      bf16x16 bk0 = ldB(Kr + trow * 64 + 16 * hi);
      bf16x16 bk1 = ldB(Kr + trow * 64 + 32 + 16 * hi);
      f32x8 acc = zero;
      acc = wmma_bf16(aq0, bk0, acc);
      acc = wmma_bf16(aq1, bk1, acc);
      st[n] = acc;
    }
    // add skewed Srel from slab, apply scale
#pragma unroll
    for (int n = 0; n < 2; ++n) {
      int t = t0 + n * 16 + lr;
#pragma unroll
      for (int i = 0; i < 8; ++i) {
        int s = s0 + wave * 16 + i + 8 * hi;
        float g;
        if (t <= s)          g = bf2f(Gl[(s - s0) * 2048 + (2047 - s + t)]);
        else if (t == s + 1) g = 0.f;
        else                 g = bf2f(Gl[(s + 1 - s0) * 2048 + (t - s - 2)]);
        st[n][i] = (st[n][i] + g) * scale;
      }
    }
    // online softmax (row reduction across the 16-lane N group)
#pragma unroll
    for (int i = 0; i < 8; ++i) {
      float mx = fmaxf(st[0][i], st[1][i]);
#pragma unroll
      for (int d = 1; d < 16; d <<= 1) mx = fmaxf(mx, __shfl_xor(mx, d, 32));
      float mnew  = fmaxf(mrow[i], mx);
      float alpha = __expf(mrow[i] - mnew);
      float p0 = __expf(st[0][i] - mnew);
      float p1 = __expf(st[1][i] - mnew);
      float ls = p0 + p1;
#pragma unroll
      for (int d = 1; d < 16; d <<= 1) ls += __shfl_xor(ls, d, 32);
      lrow[i] = lrow[i] * alpha + ls;
      mrow[i] = mnew;
      st[0][i] = p0; st[1][i] = p1;
#pragma unroll
      for (int n = 0; n < 4; ++n) Oacc[n][i] *= alpha;
    }
    // P -> LDS (bf16) for A-fragment re-layout
#pragma unroll
    for (int i = 0; i < 8; ++i) {
      int r = wave * 16 + i + 8 * hi;
      Ps[r * 32 + lr]      = f2bf(st[0][i]);
      Ps[r * 32 + 16 + lr] = f2bf(st[1][i]);
    }
    __syncthreads();
    // O += P V   (K-dim = 32, four hd n-tiles; V is raw [t][d], strided B-frag)
    bf16x16 ap = ldA(Ps + qrow * 32 + 16 * hi, Ps + qrow * 32 + 8 + 16 * hi);
#pragma unroll
    for (int n = 0; n < 4; ++n) {
      bf16x16 bv = ldBstride(Vr + (16 * hi) * 64 + n * 16 + lr, 64);
      Oacc[n] = wmma_bf16(ap, bv, Oacc[n]);
    }
    // next tile DMA must be complete and visible before buffers flip
    if (wave == 0) __builtin_amdgcn_s_wait_tensorcnt(0);
    __syncthreads();
  }

  // epilogue: O /= l, scatter to [b, s, h*64+d] fp32
  float* Og = Obuf + (size_t)b * S_LEN * DMODEL + (size_t)h * HD;
#pragma unroll
  for (int i = 0; i < 8; ++i) {
    float inv = 1.f / lrow[i];
    int s = s0 + wave * 16 + i + 8 * hi;
#pragma unroll
    for (int n = 0; n < 4; ++n)
      Og[(size_t)s * DMODEL + n * 16 + lr] = Oacc[n][i] * inv;
  }
}

// ---------------------------------------------------------------------------
// LayerNorm over last dim (1024), one block per row.
// ---------------------------------------------------------------------------
__global__ __launch_bounds__(256) void k_ln(const float* __restrict__ O,
                                            const float* __restrict__ w,
                                            const float* __restrict__ bias,
                                            float* __restrict__ out) {
  __shared__ float red[16];
  const int row = blockIdx.x;
  const float* p = O + (size_t)row * DMODEL;
  float v[4], s1 = 0.f, s2 = 0.f;
#pragma unroll
  for (int i = 0; i < 4; ++i) {
    float x = p[threadIdx.x + i * 256];
    v[i] = x; s1 += x; s2 += x * x;
  }
#pragma unroll
  for (int d = 1; d < 32; d <<= 1) {
    s1 += __shfl_xor(s1, d, 32);
    s2 += __shfl_xor(s2, d, 32);
  }
  int wv = threadIdx.x >> 5;
  if ((threadIdx.x & 31) == 0) { red[wv] = s1; red[wv + 8] = s2; }
  __syncthreads();
  s1 = 0.f; s2 = 0.f;
#pragma unroll
  for (int i = 0; i < 8; ++i) { s1 += red[i]; s2 += red[i + 8]; }
  float mu  = s1 * (1.f / DMODEL);
  float var = s2 * (1.f / DMODEL) - mu * mu;
  float inv = rsqrtf(var + 1e-5f);
#pragma unroll
  for (int i = 0; i < 4; ++i) {
    int c = threadIdx.x + i * 256;
    out[(size_t)row * DMODEL + c] = (v[i] - mu) * inv * w[c] + bias[c];
  }
}

// ---------------------------------------------------------------------------
// launch
// ---------------------------------------------------------------------------
extern "C" void kernel_launch(void* const* d_in, const int* in_sizes, int n_in,
                              void* d_out, int out_size, void* d_ws, size_t ws_size,
                              hipStream_t stream) {
  (void)in_sizes; (void)n_in; (void)out_size; (void)ws_size;
  const float* x    = (const float*)d_in[0];
  const float* Wq   = (const float*)d_in[1];
  const float* Wk   = (const float*)d_in[2];
  const float* Wv   = (const float*)d_in[3];
  const float* Er   = (const float*)d_in[4];
  const float* ln_w = (const float*)d_in[5];
  const float* ln_b = (const float*)d_in[6];
  float* out = (float*)d_out;

  // workspace layout (bytes): total 56,885,248
  char* ws = (char*)d_ws;
  unsigned short* xb   = (unsigned short*)(ws);                         //  8,388,608
  unsigned short* Wb   = (unsigned short*)(ws + 8388608);               //  6,291,456 (3x)
  unsigned short* Erb  = (unsigned short*)(ws + 14680064);              //    262,144
  unsigned short* QKVb = (unsigned short*)(ws + 14942208);              // 25,165,824 (3x)
  float*          Obuf = (float*)         (ws + 40108032);              // 16,777,216

  unsigned short* Qb = QKVb;
  unsigned short* Kb = QKVb + (size_t)2 * NH * S_LEN * HD;
  unsigned short* Vb = QKVb + (size_t)4 * NH * S_LEN * HD;

  const int NX = 2 * S_LEN * DMODEL;        // 4,194,304
  const int NW = DMODEL * DMODEL;           // 1,048,576
  k_cast<<<(NX + 255) / 256, 256, 0, stream>>>(x, xb, NX);
  k_cast<<<NW / 256, 256, 0, stream>>>(Wq, Wb, NW);
  k_cast<<<NW / 256, 256, 0, stream>>>(Wk, Wb + (size_t)NW, NW);
  k_cast<<<NW / 256, 256, 0, stream>>>(Wv, Wb + (size_t)2 * NW, NW);
  k_cast<<<(S_LEN * HD) / 256, 256, 0, stream>>>(Er, Erb, S_LEN * HD);

  k_qkv<<<dim3(DMODEL / 64, (2 * S_LEN) / 64, 3), 128, 0, stream>>>(xb, Wb, QKVb);

  const unsigned FLASH_LDS = LDS_TOTAL_HALFS * 2;   // 296,960 bytes
  (void)hipFuncSetAttribute((const void*)k_flash,
                            hipFuncAttributeMaxDynamicSharedMemorySize, (int)FLASH_LDS);
  k_flash<<<dim3(S_LEN / BQ, NH, 2), 128, FLASH_LDS, stream>>>(Qb, Kb, Vb, Erb, Obuf);

  k_ln<<<2 * S_LEN, 256, 0, stream>>>(Obuf, ln_w, ln_b, out);
}
